// SpatioConvLayerCheb_10050223473194
// MI455X (gfx1250) — matrix operand order repaired and checked by reading, not compiled, not run on hardware
//
#include <hip/hip_runtime.h>
#include <hip/hip_bf16.h>

// ---------------------------------------------------------------------------
// Problem constants (from reference setup_inputs)
// ---------------------------------------------------------------------------
#define BB   4
#define SS   12
#define NN   10000
#define DD   32
#define HH   4
#define KK   3
#define EE   80000
#define BF   (BB * SS)          // 48 fused batch
#define MROWS ((size_t)BF * NN) // 480000 GEMM rows

typedef __attribute__((ext_vector_type(16))) _Float16 v16h;
typedef __attribute__((ext_vector_type(8)))  _Float16 v8h;
typedef __attribute__((ext_vector_type(8)))  float    v8f;

// ---------------------------------------------------------------------------
// Kernel 0: one-time f32 -> f16 conversion of x (read 12x afterwards)
// ---------------------------------------------------------------------------
__global__ void cvt_f16_kernel(const float* __restrict__ x,
                               _Float16* __restrict__ xh, int n) {
    int i = blockIdx.x * blockDim.x + threadIdx.x;
    if (i < n) xh[i] = (_Float16)x[i];
}

// ---------------------------------------------------------------------------
// Kernel 1: z = x @ W[h,k] via v_wmma_f32_16x16x32_f16, fused with
//           s_s = z @ a_src[h], s_d = z @ a_dst[h].
// One wave per 16-row tile; 8 waves (256 threads) per block.
// ---------------------------------------------------------------------------
__global__ void __launch_bounds__(256)
gemm_wmma_kernel(const _Float16* __restrict__ xh,
                 const float* __restrict__ W,
                 const float* __restrict__ aSrcG,
                 const float* __restrict__ aDstG,
                 float* __restrict__ zg,
                 float* __restrict__ ss,
                 float* __restrict__ sd,
                 int h, int k) {
    __shared__ _Float16 Wt[32 * 32];   // transposed f16 W: Wt[n*32 + kk]
    __shared__ float aS[32], aD[32];
    __shared__ float zs[8 * 16 * 32];  // per-wave 16x32 f32 z staging

    const int tid  = threadIdx.x;
    const int lane = tid & 31;
    const int wave = tid >> 5;

    // Cooperative load: W[h,k] (32x32 f32, row-major [k][n]) -> transposed f16
    const float* Whk = W + ((size_t)(h * KK + k)) * 32 * 32;
    for (int i = tid; i < 1024; i += 256) {
        int kk = i >> 5, nn = i & 31;
        Wt[nn * 32 + kk] = (_Float16)Whk[i];
    }
    if (tid < 32)            aS[tid]      = aSrcG[h * 32 + tid];
    else if (tid < 64)       aD[tid - 32] = aDstG[h * 32 + (tid - 32)];
    __syncthreads();

    const size_t tile = (size_t)blockIdx.x * 8 + wave;   // 30000 tiles exactly
    const size_t row0 = tile * 16;

    // ---- A fragment: 16x32 f16. Lane l<16: M=l, K in {0..7, 16..23};
    //                              lane l>=16: M=l-16, K in {8..15, 24..31}.
    const int rowL = lane & 15;
    const int k0   = (lane >> 4) << 3;       // 0 or 8
    const _Float16* xrow = xh + (row0 + rowL) * 32;
    v8h a_lo = *(const v8h*)(xrow + k0);
    v8h a_hi = *(const v8h*)(xrow + k0 + 16);
    v16h A;
#pragma unroll
    for (int i = 0; i < 8; ++i) { A[i] = a_lo[i]; A[i + 8] = a_hi[i]; }

    const int wbase = wave * 16 * 32;
    const int kb    = (lane >> 4) << 3;      // 0 or 8 (B K-range per lane)

#pragma unroll
    for (int t = 0; t < 2; ++t) {            // N-tiles: cols 0-15, 16-31
        // ---- B fragment: 32x16 f16. Lane l<16: N=l, K in {0..7, 16..23};
        //                             lane l>=16: N=l-16, K in {8..15, 24..31}.
        const int ncol = (lane & 15) + 16 * t;
        const _Float16* wcol = &Wt[ncol * 32 + kb];
        v8h b_lo = *(const v8h*)(wcol);
        v8h b_hi = *(const v8h*)(wcol + 16);
        v16h Bm;
#pragma unroll
        for (int i = 0; i < 8; ++i) { Bm[i] = b_lo[i]; Bm[i + 8] = b_hi[i]; }

        v8f c = {};
        c = __builtin_amdgcn_wmma_f32_16x16x32_f16(
                false, A, false, Bm, (short)0, c, false, false);

        // C layout: VGPR v -> M = v + 8*(lane>=16), N = lane&15 (+16*t)
        const int mofs = (lane >> 4) << 3;
#pragma unroll
        for (int v = 0; v < 8; ++v)
            zs[wbase + (v + mofs) * 32 + ncol] = c[v];
    }
    __syncthreads();

    // ---- fused matvecs: lanes 0-15 -> s_s rows 0-15; lanes 16-31 -> s_d
    {
        const int r = lane & 15;
        const float* zr = &zs[wbase + r * 32];
        const float* av = (lane < 16) ? aS : aD;
        float dot = 0.f;
#pragma unroll
        for (int d = 0; d < 32; ++d) dot += zr[d] * av[d];
        const size_t rowg = row0 + r;
        if (lane < 16) ss[rowg] = dot; else sd[rowg] = dot;
    }

    // ---- coalesced z writeback: lane writes 16 consecutive floats
    {
        const float4* zsrc = (const float4*)&zs[wbase + lane * 16];
        float4* zdst = (float4*)(zg + row0 * 32 + (size_t)lane * 16);
#pragma unroll
        for (int j = 0; j < 4; ++j) zdst[j] = zsrc[j];
    }
}

// ---------------------------------------------------------------------------
// Kernel 2: edge logits + leaky relu + segment max (monotone-uint atomicMax)
// ---------------------------------------------------------------------------
__device__ __forceinline__ unsigned fmap(float f) {
    unsigned u = __float_as_uint(f);
    return (u & 0x80000000u) ? ~u : (u | 0x80000000u);
}
__device__ __forceinline__ float funmap(unsigned u) {
    return __uint_as_float((u & 0x80000000u) ? (u & 0x7FFFFFFFu) : ~u);
}

__global__ void edge_logit_max_kernel(const int* __restrict__ src,
                                      const int* __restrict__ dst,
                                      const float* __restrict__ ss,
                                      const float* __restrict__ sd,
                                      float* __restrict__ ebuf,
                                      unsigned* __restrict__ mbuf) {
    int e = blockIdx.x * blockDim.x + threadIdx.x;
    if (e >= EE) return;
    int b = blockIdx.y;
    int s = src[e], d = dst[e];
    float v = ss[(size_t)b * NN + s] + sd[(size_t)b * NN + d];
    v = v > 0.f ? v : 0.2f * v;                 // leaky relu
    ebuf[(size_t)b * EE + e] = v;
    atomicMax(&mbuf[(size_t)b * NN + d], fmap(v));
}

// ---------------------------------------------------------------------------
// Kernel 3: ex = exp(e - max[dst]); den[dst] += ex
// ---------------------------------------------------------------------------
__global__ void edge_exp_sum_kernel(const int* __restrict__ dst,
                                    float* __restrict__ ebuf,
                                    const unsigned* __restrict__ mbuf,
                                    float* __restrict__ den) {
    int e = blockIdx.x * blockDim.x + threadIdx.x;
    if (e >= EE) return;
    int b = blockIdx.y;
    int d = dst[e];
    float m = funmap(mbuf[(size_t)b * NN + d]);
    float ex = __expf(ebuf[(size_t)b * EE + e] - m);
    ebuf[(size_t)b * EE + e] = ex;
    atomicAdd(&den[(size_t)b * NN + d], ex);
}

// ---------------------------------------------------------------------------
// Kernel 4: acc[dst] += (ex/den * support_w) * z[src]   (32-wide, L2 atomics)
// ---------------------------------------------------------------------------
__global__ void edge_scatter_kernel(const int* __restrict__ src,
                                    const int* __restrict__ dst,
                                    const float* __restrict__ ebuf,
                                    const float* __restrict__ den,
                                    const float* __restrict__ sw,
                                    const float* __restrict__ zg,
                                    float* __restrict__ acc,
                                    int k) {
    int e = blockIdx.x * blockDim.x + threadIdx.x;
    if (e >= EE) return;
    int b = blockIdx.y;
    int s = src[e], d = dst[e];
    float alpha = ebuf[(size_t)b * EE + e]
                / (den[(size_t)b * NN + d] + 1e-9f)
                * sw[(size_t)k * EE + e];
    const float4* zp = (const float4*)(zg + ((size_t)b * NN + s) * 32);
    float* ap = acc + ((size_t)d * BF + b) * 32;
#pragma unroll
    for (int j = 0; j < 8; ++j) {
        float4 zv = zp[j];
        atomicAdd(ap + 4 * j + 0, alpha * zv.x);
        atomicAdd(ap + 4 * j + 1, alpha * zv.y);
        atomicAdd(ap + 4 * j + 2, alpha * zv.z);
        atomicAdd(ap + 4 * j + 3, alpha * zv.w);
    }
}

// ---------------------------------------------------------------------------
// Kernel 5: out = relu(acc/H + inputs)   ([n,Bf,D] -> [Bf,n,D] transpose)
// ---------------------------------------------------------------------------
__global__ void finalize_kernel(const float* __restrict__ acc,
                                const float* __restrict__ inp,
                                float* __restrict__ out, int total) {
    int i = blockIdx.x * blockDim.x + threadIdx.x;
    if (i >= total) return;
    int dd = i & 31;
    int nb = i >> 5;
    int node = nb % NN;
    int bf   = nb / NN;
    float v = acc[((size_t)node * BF + bf) * 32 + dd] * (1.0f / HH) + inp[i];
    out[i] = v > 0.f ? v : 0.f;
}

// ---------------------------------------------------------------------------
// Host side
// ---------------------------------------------------------------------------
extern "C" void kernel_launch(void* const* d_in, const int* in_sizes, int n_in,
                              void* d_out, int out_size, void* d_ws, size_t ws_size,
                              hipStream_t stream) {
    const float* x    = (const float*)d_in[0];   // [B,S,N,D]
    const int*   src  = (const int*)d_in[1];     // [E]
    const int*   dst  = (const int*)d_in[2];     // [E]
    const float* sw   = (const float*)d_in[3];   // [K,E]
    const float* W    = (const float*)d_in[4];   // [H,K,D,D]
    const float* aSrc = (const float*)d_in[5];   // [H,D]
    const float* aDst = (const float*)d_in[6];   // [H,D]
    float* out = (float*)d_out;

    // ---- workspace carve-up (all 256B aligned) ------------------------------
    const size_t nXh  = MROWS * DD;             // f16 x
    const size_t nZ   = MROWS * DD;             // f32 z
    const size_t nAcc = (size_t)NN * BF * DD;   // f32 acc
    const size_t nS   = MROWS;                  // s_s / s_d
    const size_t nE   = (size_t)BF * EE;        // edge buffer
    const size_t nM   = MROWS;                  // max / den

    char* p = (char*)d_ws;
    auto carve = [&](size_t bytes) {
        char* r = p;
        p += (bytes + 255) & ~(size_t)255;
        return r;
    };
    _Float16* xh   = (_Float16*)carve(nXh * 2);
    float*    zg   = (float*)   carve(nZ * 4);
    float*    acc  = (float*)   carve(nAcc * 4);
    float*    ss   = (float*)   carve(nS * 4);
    float*    sd   = (float*)   carve(nS * 4);
    float*    ebuf = (float*)   carve(nE * 4);
    unsigned* mbuf = (unsigned*)carve(nM * 4);
    float*    den  = (float*)   carve(nM * 4);

    // ---- one-time prep ------------------------------------------------------
    hipMemsetAsync(acc, 0, nAcc * 4, stream);
    {
        int total = (int)nXh;
        cvt_f16_kernel<<<(total + 255) / 256, 256, 0, stream>>>(x, xh, total);
    }

    const int eblocks = (EE + 255) / 256;
    const dim3 egrid(eblocks, BF);
    const int gemm_blocks = (int)(MROWS / 16 / 8);   // 3750

    // ---- per (h,k) pipeline: everything stays L2-resident -------------------
    for (int h = 0; h < HH; ++h) {
        for (int k = 0; k < KK; ++k) {
            hipMemsetAsync(mbuf, 0, nM * 4, stream);   // 0 == mapped -inf
            hipMemsetAsync(den,  0, nM * 4, stream);

            gemm_wmma_kernel<<<gemm_blocks, 256, 0, stream>>>(
                xh, W, aSrc, aDst, zg, ss, sd, h, k);

            edge_logit_max_kernel<<<egrid, 256, 0, stream>>>(
                src, dst, ss, sd, ebuf, mbuf);

            edge_exp_sum_kernel<<<egrid, 256, 0, stream>>>(
                dst, ebuf, mbuf, den);

            edge_scatter_kernel<<<egrid, 256, 0, stream>>>(
                src, dst, ebuf, den, sw, zg, acc, k);
        }
    }

    // ---- residual + relu + transpose ---------------------------------------
    {
        int total = (int)(MROWS * DD);
        finalize_kernel<<<(total + 255) / 256, 256, 0, stream>>>(acc, x, out, total);
    }
}
